// RegionProposalNetwork_42829413876200
// MI455X (gfx1250) — matrix operand order repaired and checked by reading, not compiled, not run on hardware
//
#include <hip/hip_runtime.h>
#include <hip/hip_bf16.h>
#include <hip/hip_fp16.h>
#include <stdint.h>

typedef __attribute__((ext_vector_type(16))) _Float16 v16h;
typedef __attribute__((ext_vector_type(8)))  float    v8f;
typedef __attribute__((ext_vector_type(4)))  unsigned int v4u32;
typedef __attribute__((ext_vector_type(8)))  int      v8i32;
typedef __attribute__((ext_vector_type(4)))  int      v4i32;

#define N_IMG   8
#define HF      64
#define NPIX    4096          // 64*64
#define CIN     512
#define COUT    512
#define PW      66            // padded (halo=1)
#define A_PER   9
#define NANCH   36864         // 4096*9
#define NSORT   65536
#define NPRE    2000
#define NPOST   300
#define NEG_INF (-1.0e9f)
#define NSTEP   144           // 9 taps * 16 ci-chunks of 32

// ---- d_out element offsets (reference tuple, flattened in return order) ----
#define O_LOCS   0u           // (8,36864,4) f32
#define O_SCORES 1179648u     // (8,36864,2) f32
#define O_ROIS   1769472u     // (2400,4)    f32
#define O_INDS   1779072u     // (2400,)     i32
#define O_ANCH   1781472u     // (36864,4)   f32

// ---- workspace byte offsets (all 256B aligned) ----
#define OFF_XH    ((size_t)0)                       // 8*66*66*512 f16 = 35,684,352
#define OFF_WH    ((size_t)35684352)                // 9*512*512 f16   =  4,718,592
#define OFF_HID   ((size_t)40402944)                // 8*4096*512 f16  = 33,554,432
#define OFF_BOX   ((size_t)73957376)                // 8*36864*4 f32   =  4,718,592
#define OFF_KEYS  ((size_t)78675968)                // 8*65536 u64     =  4,194,304
#define OFF_TOP   ((size_t)82870272)                // 8*2000*4 f32    =    256,000
#define OFF_K0    ((size_t)83126272)                // 8*2000 u32      =     64,000
#define OFF_MASK  ((size_t)83190272)                // 8*2000*64 u32   =  4,096,000
#define OFF_KEEP  ((size_t)87286272)                // 8*2000 u32      =     64,000

// ============================================================
// Pack input NCHW f32 -> padded NHWC f16 (zero halo)
// ============================================================
__global__ void k_pack_x(const float* __restrict__ x, _Float16* __restrict__ xh) {
  size_t id = (size_t)blockIdx.x * blockDim.x + threadIdx.x;
  if (id >= (size_t)N_IMG * PW * PW * CIN) return;
  int c = (int)(id & (CIN - 1));
  size_t p = id >> 9;
  int w = (int)(p % PW); p /= PW;
  int h = (int)(p % PW);
  int n = (int)(p / PW);
  float v = 0.f;
  if (h >= 1 && h <= HF && w >= 1 && w <= HF)
    v = x[(((size_t)n * CIN + c) * HF + (size_t)(h - 1)) * HF + (size_t)(w - 1)];
  xh[id] = (_Float16)v;
}

// Pack conv weights [co][ci][3][3] f32 -> [g][co][ci] f16 (B fragments contiguous in K)
__global__ void k_pack_w(const float* __restrict__ cw, _Float16* __restrict__ wh) {
  int id = blockIdx.x * blockDim.x + threadIdx.x;
  if (id >= 9 * COUT * CIN) return;
  int ci = id & (CIN - 1);
  int co = (id >> 9) & (COUT - 1);
  int g  = id >> 18;
  wh[id] = (_Float16)cw[((size_t)co * CIN + ci) * 9 + (size_t)g];
}

// ============================================================
// 3x3 conv as implicit GEMM via v_wmma_f32_16x16x32_f16.
// Block = 8 waves (256 threads) = one 16(M) x 512(N) tile.
// Double-buffered TDM pipeline: wave 0 DMAs the next 512x32 f16 weight
// slab (32 KB) into LDS buf[cur^1] while all 8 waves consume buf[cur]
// via ds_load_b128. One s_wait_tensorcnt + one WG barrier per K-step.
// ============================================================
__global__ __launch_bounds__(256) void k_conv_wmma(
    const _Float16* __restrict__ xh, const _Float16* __restrict__ wh,
    const float* __restrict__ bias, _Float16* __restrict__ hidden) {
  __shared__ _Float16 bsm[2][COUT * 32];   // 2 x 32 KB staged B tiles: [co][ci32]

  const int lane = threadIdx.x & 31;
  const int ng   = threadIdx.x >> 5;       // wave-in-block = 64-channel group
  const int mt   = blockIdx.x;             // 2048 M tiles of 16 rows
  const int m0 = mt << 4;
  const int n  = m0 >> 12;
  const int pix = m0 & (NPIX - 1);
  const int ho = pix >> 6;
  const int wo = pix & 63;                 // multiple of 16 -> tile stays in one row
  const int mrow = lane & 15;
  const int akb = (lane < 16) ? 0 : 8;     // A-frag K base (ISA 16-bit A layout)
  const int bkb = (lane < 16) ? 0 : 8;     // B-frag K base (ISA 16-bit B layout)
  const int bcol = lane & 15;
  const int wo_l = wo + mrow;

  v8f acc0 = {}, acc1 = {}, acc2 = {}, acc3 = {};
  const int co0 = (ng << 6) + bcol;
  const unsigned long long whb = (unsigned long long)(size_t)wh;

  // ---- issue one 512x32 weight-slab DMA into LDS buffer `buf` ----
#define ISSUE_TDM(STEP, BUF)                                                 \
  {                                                                          \
    const int g_ = (STEP) >> 4;                                              \
    const int ci_ = ((STEP) & 15) << 5;                                      \
    const unsigned long long ga =                                            \
        whb + (((unsigned long long)g_ * COUT * CIN) +                       \
               (unsigned long long)ci_) * 2ull;                              \
    v4u32 d0;                                                                \
    d0.x = 1u;                                     /* count=1, user mode  */ \
    d0.y = (unsigned)(BUF) * (COUT * 32u * 2u);    /* lds_addr            */ \
    d0.z = (unsigned)(ga & 0xFFFFFFFFull);         /* global_addr[31:0]   */ \
    d0.w = (unsigned)((ga >> 32) & 0x1FFFFFFull)   /* global_addr[56:32]  */ \
           | (2u << 30);                           /* type = 2 ("image")  */ \
    v8i32 d1;                                                                \
    d1[0] = (int)(1u << 16);      /* wg_mask=0, data_size=1 (2 bytes)     */ \
    d1[1] = (int)(512u << 16);    /* tensor_dim0[15:0]=512 @bits[63:48]   */ \
    d1[2] = (int)(4608u << 16);   /* dim0[31:16]=0, tensor_dim1[15:0]     */ \
    d1[3] = (int)(32u << 16);     /* dim1[31:16]=0, tile_dim0=32          */ \
    d1[4] = 512;                  /* tile_dim1=512, tile_dim2=0           */ \
    d1[5] = 512;                  /* tensor_dim0_stride[31:0]=512         */ \
    d1[6] = 0;                                                               \
    d1[7] = 0;                                                               \
    v4i32 d2 = {0, 0, 0, 0};                                                 \
    v4i32 d3 = {0, 0, 0, 0};                                                 \
    v8i32 d4 = {0, 0, 0, 0, 0, 0, 0, 0};                                     \
    __builtin_amdgcn_tensor_load_to_lds(d0, d1, d2, d3, d4, 0);              \
  }

  if (ng == 0) ISSUE_TDM(0, 0)             // prologue: fill buffer 0

  for (int step = 0; step < NSTEP; ++step) {
    const int g  = step >> 4;
    const int ci = (step & 15) << 5;
    const int cur = step & 1;
    const int kh = g / 3, kw = g - kh * 3;

    if (ng == 0) __builtin_amdgcn_s_wait_tensorcnt(0);  // buf[cur] filled
    __syncthreads();                     // consumers released; buf[cur^1] free
    asm volatile("" ::: "memory");       // LDS contents changed: no load CSE

    if (ng == 0 && step + 1 < NSTEP) ISSUE_TDM(step + 1, cur ^ 1)

    union { v16h v; uint4 q[2]; } A, B0, B1, B2, B3;
    // A lane holds K {akb..akb+7, akb+16..akb+23} -> two contiguous b128 loads
    const _Float16* ap = xh + ((((size_t)n * PW) + (size_t)(ho + kh)) * PW +
                               (size_t)(wo_l + kw)) * CIN + (size_t)(akb + ci);
    A.q[0] = *(const uint4*)(ap);
    A.q[1] = *(const uint4*)(ap + 16);
    // B lane holds K {bkb..bkb+7, bkb+16..bkb+23} in LDS row co -> ds_load_b128 x2
    const _Float16* b0 = &bsm[cur][0] + ((size_t)co0 << 5) + bkb;
    const _Float16* b1 = b0 + (16 << 5);
    const _Float16* b2 = b0 + (32 << 5);
    const _Float16* b3 = b0 + (48 << 5);
    B0.q[0] = *(const uint4*)(b0);  B0.q[1] = *(const uint4*)(b0 + 16);
    B1.q[0] = *(const uint4*)(b1);  B1.q[1] = *(const uint4*)(b1 + 16);
    B2.q[0] = *(const uint4*)(b2);  B2.q[1] = *(const uint4*)(b2 + 16);
    B3.q[0] = *(const uint4*)(b3);  B3.q[1] = *(const uint4*)(b3 + 16);
    acc0 = __builtin_amdgcn_wmma_f32_16x16x32_f16(false, A.v, false, B0.v, (short)0, acc0, false, false);
    acc1 = __builtin_amdgcn_wmma_f32_16x16x32_f16(false, A.v, false, B1.v, (short)0, acc1, false, false);
    acc2 = __builtin_amdgcn_wmma_f32_16x16x32_f16(false, A.v, false, B2.v, (short)0, acc2, false, false);
    acc3 = __builtin_amdgcn_wmma_f32_16x16x32_f16(false, A.v, false, B3.v, (short)0, acc3, false, false);
  }
#undef ISSUE_TDM

  // C/D layout: VGPR r -> M = m0 + r + (lane<16?0:8), N = lane&15
  const int mlo = (lane < 16) ? 0 : 8;
#define STORE_TILE(ACC, S)                                                  \
  {                                                                         \
    const int co = (ng << 6) + ((S) << 4) + bcol;                           \
    const float bb = bias[co];                                              \
    _Pragma("unroll")                                                       \
    for (int r = 0; r < 8; ++r) {                                           \
      float v = ACC[r] + bb;                                                \
      v = fmaxf(v, 0.f);                                                    \
      hidden[((size_t)(m0 + r + mlo)) * COUT + (size_t)co] = (_Float16)v;   \
    }                                                                       \
  }
  STORE_TILE(acc0, 0)
  STORE_TILE(acc1, 1)
  STORE_TILE(acc2, 2)
  STORE_TILE(acc3, 3)
#undef STORE_TILE
}

// ============================================================
// 1x1 score/loc heads: write rpn_scores / rpn_locs directly to d_out
// ============================================================
__global__ void k_heads(const _Float16* __restrict__ hidden,
                        const float* __restrict__ sw, const float* __restrict__ sb,
                        const float* __restrict__ lw, const float* __restrict__ lb,
                        float* __restrict__ out_locs, float* __restrict__ out_scores) {
  int id = blockIdx.x * blockDim.x + threadIdx.x;
  if (id >= N_IMG * NPIX * 54) return;
  int o = id % 54;
  int p = id / 54;                       // n*4096 + pix
  int pix = p & (NPIX - 1);
  int n = p >> 12;
  const _Float16* h = hidden + (size_t)p * CIN;
  const float* wv;
  float acc;
  if (o < 18) { wv = sw + (size_t)o * CIN;        acc = sb[o]; }
  else        { wv = lw + (size_t)(o - 18) * CIN; acc = lb[o - 18]; }
#pragma unroll 8
  for (int c = 0; c < CIN; ++c) acc += (float)h[c] * wv[c];
  if (o < 18) {
    int a = o >> 1, s = o & 1;
    out_scores[((size_t)n * NANCH + (size_t)pix * A_PER + a) * 2 + s] = acc;
  } else {
    int oo = o - 18, a = oo >> 2, c4 = oo & 3;
    out_locs[((size_t)n * NANCH + (size_t)pix * A_PER + a) * 4 + c4] = acc;
  }
}

// ============================================================
// Anchors (written to d_out anchors segment)
// ============================================================
__global__ void k_anchors(float* __restrict__ anch) {
  int i = blockIdx.x * blockDim.x + threadIdx.x;
  if (i >= NANCH) return;
  int a = i % 9, pix = i / 9;
  int ho = pix >> 6, wo = pix & 63;
  float sy = (float)(ho * 16), sx = (float)(wo * 16);
  const float ratios[3] = {0.5f, 1.f, 2.f};
  const float scales[3] = {8.f, 16.f, 32.f};
  int ri = a / 3, si = a % 3;
  float hh = 16.f * scales[si] * sqrtf(ratios[ri]);
  float ww = 16.f * scales[si] * sqrtf(1.f / ratios[ri]);
  float* o = anch + (size_t)i * 4;
  o[0] = sy + 8.f - 0.5f * hh;
  o[1] = sx + 8.f - 0.5f * ww;
  o[2] = sy + 8.f + 0.5f * hh;
  o[3] = sx + 8.f + 0.5f * ww;
}

// ============================================================
// Decode boxes, clip, min-size filter, build sortable keys.
// key = ordered(score)<<32 | (0xFFFFFFFF - i)  -> descending sort
// gives top_k semantics (value desc, ties by lower index).
// ============================================================
__global__ void k_decode(const float* __restrict__ out_locs,
                         const float* __restrict__ out_scores,
                         const float* __restrict__ anch,
                         const int* __restrict__ ihp, const int* __restrict__ iwp,
                         float* __restrict__ boxes,
                         unsigned long long* __restrict__ keys) {
  int id = blockIdx.x * blockDim.x + threadIdx.x;     // 8 * 65536
  int n = id >> 16;
  int i = id & (NSORT - 1);
  if (n >= N_IMG) return;
  unsigned long long* kp = keys + (size_t)n * NSORT;
  if (i >= NANCH) { kp[i] = 0ull; return; }           // padding: smallest key
  float fh = (float)ihp[0], fw = (float)iwp[0];
  const float* a = anch + (size_t)i * 4;
  const float* l = out_locs + ((size_t)n * NANCH + i) * 4;
  float ah = a[2] - a[0], aw = a[3] - a[1];
  float acy = a[0] + 0.5f * ah, acx = a[1] + 0.5f * aw;
  float cy = l[0] * ah + acy, cx = l[1] * aw + acx;
  float hh = expf(l[2]) * ah, ww = expf(l[3]) * aw;
  float y1 = fminf(fmaxf(cy - 0.5f * hh, 0.f), fh);
  float x1 = fminf(fmaxf(cx - 0.5f * ww, 0.f), fw);
  float y2 = fminf(fmaxf(cy + 0.5f * hh, 0.f), fh);
  float x2 = fminf(fmaxf(cx + 0.5f * ww, 0.f), fw);
  float* b = boxes + ((size_t)n * NANCH + i) * 4;
  b[0] = y1; b[1] = x1; b[2] = y2; b[3] = x2;
  const float* s = out_scores + ((size_t)n * NANCH + i) * 2;
  float fg = 1.f / (1.f + expf(s[0] - s[1]));         // softmax [...,1]
  bool ok = ((y2 - y1) >= 16.f) && ((x2 - x1) >= 16.f);
  float sc = ok ? fg : NEG_INF;
  unsigned u = __float_as_uint(sc);
  u = (u & 0x80000000u) ? ~u : (u | 0x80000000u);     // order-preserving map
  kp[i] = ((unsigned long long)u << 32) | (unsigned long long)(0xFFFFFFFFu - (unsigned)i);
}

// ============================================================
// Per-image bitonic sort of 65536 keys, descending. One WG/image.
// ============================================================
__global__ __launch_bounds__(1024) void k_bitonic(unsigned long long* __restrict__ keys) {
  unsigned long long* k = keys + (size_t)blockIdx.x * NSORT;
  for (int kk = 2; kk <= NSORT; kk <<= 1) {
    for (int j = kk >> 1; j > 0; j >>= 1) {
      __syncthreads();
      for (int t = threadIdx.x; t < NSORT; t += 1024) {
        int l = t ^ j;
        if (l > t) {
          unsigned long long a = k[t], b = k[l];
          bool up = ((t & kk) == 0);
          bool sw = up ? (a < b) : (a > b);             // descending overall
          if (sw) { k[t] = b; k[l] = a; }
        }
      }
    }
  }
}

// Gather top-2000 boxes + keep0 flags
__global__ void k_gather(const unsigned long long* __restrict__ keys,
                         const float* __restrict__ boxes,
                         float* __restrict__ topbox, unsigned* __restrict__ keep0) {
  int id = blockIdx.x * blockDim.x + threadIdx.x;
  if (id >= N_IMG * NPRE) return;
  int n = id / NPRE, r = id % NPRE;
  unsigned long long key = keys[(size_t)n * NSORT + r];
  unsigned idx = 0xFFFFFFFFu - (unsigned)(key & 0xFFFFFFFFull);
  unsigned u = (unsigned)(key >> 32);
  unsigned orig = (u & 0x80000000u) ? (u & 0x7FFFFFFFu) : ~u;
  float sc = __uint_as_float(orig);
  float b0 = 0.f, b1 = 0.f, b2 = 0.f, b3 = 0.f;
  unsigned k0 = 0u;
  if (idx < NANCH && sc > NEG_INF * 0.5f) {
    const float* b = boxes + ((size_t)n * NANCH + idx) * 4;
    b0 = b[0]; b1 = b[1]; b2 = b[2]; b3 = b[3];
    k0 = 1u;
  }
  float* t = topbox + ((size_t)n * NPRE + r) * 4;
  t[0] = b0; t[1] = b1; t[2] = b2; t[3] = b3;
  keep0[n * NPRE + r] = k0;
}

// NMS suppression bitmask: mask[n][i][w] bit t set if j=w*32+t > i and IoU>0.7
__global__ __launch_bounds__(64) void k_mask(const float* __restrict__ topbox,
                                             unsigned* __restrict__ mask) {
  const int i = blockIdx.x, n = blockIdx.y, w = threadIdx.x;
  const float* bi = topbox + ((size_t)n * NPRE + i) * 4;
  float y1 = bi[0], x1 = bi[1], y2 = bi[2], x2 = bi[3];
  float ai = (y2 - y1) * (x2 - x1);
  unsigned bits = 0u;
  int j0 = w << 5;
  for (int t = 0; t < 32; ++t) {
    int j = j0 + t;
    if (j < NPRE && j > i) {
      const float* bj = topbox + ((size_t)n * NPRE + j) * 4;
      float ty = fmaxf(y1, bj[0]), tx = fmaxf(x1, bj[1]);
      float by = fminf(y2, bj[2]), bx = fminf(x2, bj[3]);
      float ih = fmaxf(by - ty, 0.f), iw = fmaxf(bx - tx, 0.f);
      float inter = ih * iw;
      float aj = (bj[2] - bj[0]) * (bj[3] - bj[1]);
      float iou = inter / (ai + aj - inter + 1e-9f);
      if (iou > 0.7f) bits |= (1u << t);
    }
  }
  mask[((size_t)n * NPRE + i) * 64 + w] = bits;
}

// Sequential greedy NMS scan (one WG/image, words parallel across threads)
__global__ __launch_bounds__(64) void k_nms_scan(const unsigned* __restrict__ mask,
                                                 const unsigned* __restrict__ keep0,
                                                 unsigned* __restrict__ keep) {
  __shared__ unsigned rem[64];
  __shared__ int dec;
  const int n = blockIdx.x, tid = threadIdx.x;
  rem[tid] = 0u;
  __syncthreads();
  for (int i = 0; i < NPRE; ++i) {
    if (tid == 0) {
      bool sup = (rem[i >> 5] >> (i & 31)) & 1u;
      dec = (keep0[n * NPRE + i] && !sup) ? 1 : 0;
      keep[n * NPRE + i] = (unsigned)dec;
    }
    __syncthreads();
    if (dec) rem[tid] |= mask[((size_t)n * NPRE + i) * 64 + tid];
    __syncthreads();
  }
}

// Compact first 300 kept boxes (score order), zero-fill rest, write roi indices
__global__ void k_final(const float* __restrict__ topbox,
                        const unsigned* __restrict__ keep,
                        float* __restrict__ rois, int* __restrict__ inds) {
  int n = blockIdx.x;
  int cnt = 0;
  for (int i = 0; i < NPRE && cnt < NPOST; ++i) {
    if (keep[n * NPRE + i]) {
      const float* b = topbox + ((size_t)n * NPRE + i) * 4;
      float* r = rois + ((size_t)n * NPOST + cnt) * 4;
      r[0] = b[0]; r[1] = b[1]; r[2] = b[2]; r[3] = b[3];
      ++cnt;
    }
  }
  for (; cnt < NPOST; ++cnt) {
    float* r = rois + ((size_t)n * NPOST + cnt) * 4;
    r[0] = 0.f; r[1] = 0.f; r[2] = 0.f; r[3] = 0.f;
  }
  for (int r2 = 0; r2 < NPOST; ++r2) inds[n * NPOST + r2] = n;
}

// ============================================================
extern "C" void kernel_launch(void* const* d_in, const int* in_sizes, int n_in,
                              void* d_out, int out_size, void* d_ws, size_t ws_size,
                              hipStream_t stream) {
  (void)in_sizes; (void)n_in; (void)out_size; (void)ws_size;
  const float* x  = (const float*)d_in[0];
  const float* cw = (const float*)d_in[1];
  const float* cb = (const float*)d_in[2];
  const float* sw = (const float*)d_in[3];
  const float* sb = (const float*)d_in[4];
  const float* lw = (const float*)d_in[5];
  const float* lb = (const float*)d_in[6];
  const int*   ih = (const int*)d_in[7];
  const int*   iw = (const int*)d_in[8];

  float* out = (float*)d_out;
  char*  ws  = (char*)d_ws;

  _Float16* xh  = (_Float16*)(ws + OFF_XH);
  _Float16* wh  = (_Float16*)(ws + OFF_WH);
  _Float16* hid = (_Float16*)(ws + OFF_HID);
  float*    box = (float*)(ws + OFF_BOX);
  unsigned long long* keys = (unsigned long long*)(ws + OFF_KEYS);
  float*    top = (float*)(ws + OFF_TOP);
  unsigned* k0  = (unsigned*)(ws + OFF_K0);
  unsigned* msk = (unsigned*)(ws + OFF_MASK);
  unsigned* kp  = (unsigned*)(ws + OFF_KEEP);

  float* out_locs   = out + O_LOCS;
  float* out_scores = out + O_SCORES;
  float* rois       = out + O_ROIS;
  int*   inds       = (int*)(out + O_INDS);
  float* anch       = out + O_ANCH;

  // 1) pack input (with halo) and weights to f16
  {
    size_t tot = (size_t)N_IMG * PW * PW * CIN;
    k_pack_x<<<(unsigned)((tot + 255) / 256), 256, 0, stream>>>(x, xh);
  }
  k_pack_w<<<(9 * COUT * CIN) / 256, 256, 0, stream>>>(cw, wh);
  k_anchors<<<(NANCH + 255) / 256, 256, 0, stream>>>(anch);

  // 2) 3x3 conv + bias + ReLU via WMMA + double-buffered TDM weight staging
  //    2048 blocks x 8 waves; block = 16 rows x all 512 channels
  k_conv_wmma<<<2048, 256, 0, stream>>>(xh, wh, cb, hid);

  // 3) 1x1 heads -> rpn_locs / rpn_scores directly in d_out
  k_heads<<<(N_IMG * NPIX * 54) / 256, 256, 0, stream>>>(hid, sw, sb, lw, lb,
                                                         out_locs, out_scores);

  // 4) decode boxes + sort keys
  k_decode<<<(N_IMG * NSORT) / 256, 256, 0, stream>>>(out_locs, out_scores, anch,
                                                      ih, iw, box, keys);

  // 5) per-image descending bitonic sort (top_k equivalent)
  k_bitonic<<<N_IMG, 1024, 0, stream>>>(keys);

  // 6) gather top-2000 + NMS
  k_gather<<<(N_IMG * NPRE + 255) / 256, 256, 0, stream>>>(keys, box, top, k0);
  k_mask<<<dim3(NPRE, N_IMG), 64, 0, stream>>>(top, msk);
  k_nms_scan<<<N_IMG, 64, 0, stream>>>(msk, k0, kp);

  // 7) compact rois + indices
  k_final<<<N_IMG, 1, 0, stream>>>(top, kp, rois, inds);
}
